// deepRetinotopy_OptionB_29326036697705
// MI455X (gfx1250) — matrix operand order repaired and checked by reading, not compiled, not run on hardware
//
#include <hip/hip_runtime.h>
#include <math.h>

typedef __attribute__((ext_vector_type(16))) _Float16 v16h;
typedef __attribute__((ext_vector_type(8)))  float    v8f;

#define WAVE 32

__device__ __forceinline__ float gelu_f(float x) {
    return 0.5f * x * (1.0f + erff(x * 0.70710678118654752440f));
}
__device__ __forceinline__ float elu_f(float x) {
    return x > 0.0f ? x : expm1f(x);
}
__device__ __forceinline__ float wred_sum(float v) {
    #pragma unroll
    for (int o = 16; o; o >>= 1) v += __shfl_xor(v, o, WAVE);
    return v;
}
__device__ __forceinline__ float wred_max(float v) {
    #pragma unroll
    for (int o = 16; o; o >>= 1) v = fmaxf(v, __shfl_xor(v, o, WAVE));
    return v;
}

// ---------------------------------------------------------------------------
// Degree counts: cnt[t] = in-degree (tgt), degS[s] = out-degree (src)
// ---------------------------------------------------------------------------
__global__ void k_count_deg(const int* __restrict__ src, const int* __restrict__ tgt,
                            float* __restrict__ cnt, float* __restrict__ degS, int E) {
    int e = blockIdx.x * blockDim.x + threadIdx.x;
    if (e >= E) return;
    atomicAdd(&cnt[tgt[e]], 1.0f);
    atomicAdd(&degS[src[e]], 1.0f);
}

// ---------------------------------------------------------------------------
// Edge features: avg dist to 5 NN (excluding self), out-degree, density.
// One wave per node; per-lane top-6 insertion sort, LDS merge by lane 0.
// ---------------------------------------------------------------------------
__global__ void k_edge_feats(const float* __restrict__ pos, const float* __restrict__ degS,
                             float* __restrict__ ef, int N) {
    __shared__ float sm[8 * 192];
    int wslot = threadIdx.x >> 5;
    int lane  = threadIdx.x & 31;
    int i = blockIdx.x * 8 + wslot;
    float t[6];
    #pragma unroll
    for (int k = 0; k < 6; ++k) t[k] = 3.0e38f;
    if (i < N) {
        float px = pos[i * 3 + 0], py = pos[i * 3 + 1], pz = pos[i * 3 + 2];
        float si = px * px + py * py + pz * pz;
        for (int j = lane; j < N; j += WAVE) {
            float qx = pos[j * 3 + 0], qy = pos[j * 3 + 1], qz = pos[j * 3 + 2];
            float sj = qx * qx + qy * qy + qz * qz;
            float d2 = si + sj - 2.0f * (px * qx + py * qy + pz * qz);
            float d = sqrtf(fmaxf(d2, 1e-12f));
            if (d < t[5]) {
                t[5] = d;
                #pragma unroll
                for (int k = 5; k > 0; --k)
                    if (t[k] < t[k - 1]) { float tm = t[k]; t[k] = t[k - 1]; t[k - 1] = tm; }
            }
        }
    }
    #pragma unroll
    for (int k = 0; k < 6; ++k) sm[wslot * 192 + lane * 6 + k] = t[k];
    __syncthreads();
    if (i < N && lane == 0) {
        float* base = &sm[wslot * 192];
        float mins[6];
        for (int m = 0; m < 6; ++m) {
            float best = 3.0e38f; int bi = 0;
            for (int k = 0; k < 192; ++k)
                if (base[k] < best) { best = base[k]; bi = k; }
            mins[m] = best; base[bi] = 3.0e38f;
        }
        // mins[0] is self-distance; average the next 5
        float avg = (mins[1] + mins[2] + mins[3] + mins[4] + mins[5]) * 0.2f;
        ef[i * 3 + 0] = avg;
        ef[i * 3 + 1] = degS[i];
        ef[i * 3 + 2] = 1.0f / (avg + 1e-6f);
    }
}

// ---------------------------------------------------------------------------
// Edge-feature encoder: 3 -> 8 (gelu) -> 4
// ---------------------------------------------------------------------------
__global__ void k_encoder(const float* __restrict__ ef,
                          const float* __restrict__ w1, const float* __restrict__ b1,
                          const float* __restrict__ w2, const float* __restrict__ b2,
                          float* __restrict__ enc, int N) {
    int n = blockIdx.x * blockDim.x + threadIdx.x;
    if (n >= N) return;
    float e0 = ef[n * 3], e1 = ef[n * 3 + 1], e2 = ef[n * 3 + 2];
    float h[8];
    #pragma unroll
    for (int j = 0; j < 8; ++j)
        h[j] = gelu_f(e0 * w1[0 * 8 + j] + e1 * w1[1 * 8 + j] + e2 * w1[2 * 8 + j] + b1[j]);
    #pragma unroll
    for (int c = 0; c < 4; ++c) {
        float v = b2[c];
        #pragma unroll
        for (int j = 0; j < 8; ++j) v = fmaf(h[j], w2[j * 4 + c], v);
        enc[n * 4 + c] = v;
    }
}

// ---------------------------------------------------------------------------
// SplineConv edge kernel: one wave per edge, lane = output channel.
// msg = sum over 8 deg-1 B-spline supports of b_s * (x_j @ W[wi_s]),
// atomically accumulated into agg[tgt].
// ---------------------------------------------------------------------------
__global__ void k_spline_edge(const float* __restrict__ h, const float* __restrict__ pseudo,
                              const int* __restrict__ src, const int* __restrict__ tgt,
                              const float* __restrict__ W, float* __restrict__ agg,
                              int E, int Cin, int Cout) {
    int wid  = (blockIdx.x * blockDim.x + threadIdx.x) >> 5;
    int lane = threadIdx.x & 31;
    if (wid >= E) return;
    int sN = src[wid], tN = tgt[wid];
    float fr[3]; int i0[3];
    #pragma unroll
    for (int d = 0; d < 3; ++d) {
        float v = pseudo[wid * 3 + d] * 24.0f;   // (K-1) = 24
        float fl = floorf(v);
        fr[d] = v - fl;
        int ii = (int)fl;
        ii = ii < 0 ? 0 : (ii > 23 ? 23 : ii);
        i0[d] = ii;
    }
    float xv = (lane < Cin) ? h[(size_t)sN * Cin + lane] : 0.0f;
    int lo = lane < Cout ? lane : 0;
    float acc = 0.0f;
    #pragma unroll
    for (int s = 0; s < 8; ++s) {
        int b0 = s & 1, b1 = (s >> 1) & 1, b2 = (s >> 2) & 1;
        float w0 = b0 ? fr[0] : 1.0f - fr[0];
        float w1 = b1 ? fr[1] : 1.0f - fr[1];
        float w2 = b2 ? fr[2] : 1.0f - fr[2];
        float bp = w0 * w1 * w2;
        int wi = (i0[0] + b0) * 625 + (i0[1] + b1) * 25 + (i0[2] + b2);
        const float* Wb = W + (size_t)wi * Cin * Cout + lo;
        float part = 0.0f;
        for (int i = 0; i < Cin; ++i) {
            float xi = __shfl(xv, i, WAVE);
            part = fmaf(xi, Wb[(size_t)i * Cout], part);
        }
        acc = fmaf(bp, part, acc);
    }
    if (lane < Cout) atomicAdd(&agg[(size_t)tN * Cout + lane], acc);
}

// ---------------------------------------------------------------------------
// SplineConv node update: h = BN(elu(agg*deg_inv + h@root + bias)); BN optional.
// ---------------------------------------------------------------------------
__global__ void k_node_update(const float* __restrict__ hin, const float* __restrict__ agg,
                              const float* __restrict__ cnt,
                              const float* __restrict__ root, const float* __restrict__ bias,
                              const float* __restrict__ bn_g, const float* __restrict__ bn_b,
                              const float* __restrict__ bn_m, const float* __restrict__ bn_v,
                              float* __restrict__ out, int N, int Cin, int Cout) {
    int t = blockIdx.x * blockDim.x + threadIdx.x;
    if (t >= N * Cout) return;
    int n = t / Cout, o = t - n * Cout;
    float di = 1.0f / fmaxf(cnt[n], 1.0f);
    float v = agg[t] * di + bias[o];
    for (int i = 0; i < Cin; ++i)
        v = fmaf(hin[(size_t)n * Cin + i], root[(size_t)i * Cout + o], v);
    v = elu_f(v);
    if (bn_g) v = (v - bn_m[o]) * (bn_g[o] * rsqrtf(bn_v[o] + 1e-5f)) + bn_b[o];
    out[t] = v;
}

// ---------------------------------------------------------------------------
// hb = h + enc @ ep_w + ep_b   (enc: [N,4], ep_w: [4,32])
// ---------------------------------------------------------------------------
__global__ void k_add_ep(const float* __restrict__ h, const float* __restrict__ enc,
                         const float* __restrict__ epw, const float* __restrict__ epb,
                         float* __restrict__ hb, int N) {
    int t = blockIdx.x * blockDim.x + threadIdx.x;
    if (t >= N * 32) return;
    int n = t >> 5, o = t & 31;
    float v = h[t] + epb[o];
    #pragma unroll
    for (int k = 0; k < 4; ++k) v = fmaf(enc[n * 4 + k], epw[k * 32 + o], v);
    hb[t] = v;
}

// ---------------------------------------------------------------------------
// LayerNorm over last dim (32)
// ---------------------------------------------------------------------------
__global__ void k_layernorm(const float* __restrict__ in, const float* __restrict__ g,
                            const float* __restrict__ b, float* __restrict__ out, int N) {
    int n = blockIdx.x * blockDim.x + threadIdx.x;
    if (n >= N) return;
    const float* r = in + (size_t)n * 32;
    float mu = 0.0f;
    #pragma unroll
    for (int c = 0; c < 32; ++c) mu += r[c];
    mu *= (1.0f / 32.0f);
    float var = 0.0f;
    #pragma unroll
    for (int c = 0; c < 32; ++c) { float d = r[c] - mu; var = fmaf(d, d, var); }
    var *= (1.0f / 32.0f);
    float inv = rsqrtf(var + 1e-5f);
    #pragma unroll
    for (int c = 0; c < 32; ++c)
        out[(size_t)n * 32 + c] = (r[c] - mu) * inv * g[c] + b[c];
}

// ===========================================================================
// WMMA path: A and B are pre-packed into the exact per-lane register layout of
// V_WMMA_F32_16X16X32_F16 so the matmul kernel issues only wide b128 loads.
//
// Packed layout (halves): p[((tile * Kblk + kblk) * 32 + lane) * 16 + j]
//   A: lane -> row M = tile_m*16 + (lane&15); j -> K per ISA A table
//      (half = lane>>4: j<8 -> K = 8*half + j ; j>=8 -> K = 16 + 8*half + (j-8))
//   B: lane -> col N = tile_n*16 + (lane&15); j -> K same as A table
// C/D layout: VGPR r -> M = r + 8*(lane>=16), N = lane&15.
// ===========================================================================
__global__ void k_pack_a(const float* __restrict__ A, _Float16* __restrict__ pA,
                         int M, int K, int total) {
    int t = blockIdx.x * blockDim.x + threadIdx.x;
    if (t >= total) return;
    int j    = t & 15;
    int lane = (t >> 4) & 31;
    int blk  = t >> 9;                 // tile_m * Kblk + kblk
    int Kblk = K >> 5;
    int kblk = blk % Kblk;
    int tile_m = blk / Kblk;
    int half = lane >> 4;
    int m = tile_m * 16 + (lane & 15);
    int ka = kblk * 32 + ((j < 8) ? (half * 8 + j) : (16 + half * 8 + (j - 8)));
    pA[t] = (m < M) ? (_Float16)A[(size_t)m * K + ka] : (_Float16)0.0f;
}

__global__ void k_pack_b(const float* __restrict__ Wt, _Float16* __restrict__ pB,
                         int K, int Nc, int total) {
    int t = blockIdx.x * blockDim.x + threadIdx.x;
    if (t >= total) return;
    int j    = t & 15;
    int lane = (t >> 4) & 31;
    int blk  = t >> 9;                 // tile_n * Kblk + kblk
    int Kblk = K >> 5;
    int kblk = blk % Kblk;
    int tile_n = blk / Kblk;
    int n  = tile_n * 16 + (lane & 15);
    int kb = kblk * 32 + ((j < 8) ? ((lane >> 4) * 8 + j) : (16 + (lane >> 4) * 8 + (j - 8)));
    pB[t] = (_Float16)Wt[(size_t)kb * Nc + n];
}

__global__ void k_wmma_linear(const _Float16* __restrict__ pA, const _Float16* __restrict__ pB,
                              const float* __restrict__ bias, const float* __restrict__ res,
                              float* __restrict__ out, int M, int Kblk, int Nc, int act) {
    int tile_m = blockIdx.x * 8 + (threadIdx.x >> 5);
    int Mt = (M + 15) >> 4;
    if (tile_m >= Mt) return;           // wave-uniform exit: EXEC stays full
    int tile_n = blockIdx.y;
    int lane = threadIdx.x & 31;
    const v16h* Av = (const v16h*)pA;
    const v16h* Bv = (const v16h*)pB;
    size_t abase = ((size_t)tile_m * Kblk) * 32 + lane;
    size_t bbase = ((size_t)tile_n * Kblk) * 32 + lane;
    v8f acc = {};
    #pragma unroll 4
    for (int kb = 0; kb < Kblk; ++kb) {
        v16h a = Av[abase + (size_t)kb * 32];
        v16h b = Bv[bbase + (size_t)kb * 32];
        acc = __builtin_amdgcn_wmma_f32_16x16x32_f16(
            /*neg_a=*/false, a, /*neg_b=*/false, b,
            /*c_mod=*/(short)0, acc, /*reuse_a=*/false, /*reuse_b=*/false);
    }
    int half = lane >> 4;
    int l15  = lane & 15;
    int No = tile_n * 16 + l15;
    float bs = bias[No];
    #pragma unroll
    for (int r = 0; r < 8; ++r) {
        int Mo = tile_m * 16 + r + 8 * half;
        if (Mo >= M) continue;
        float v = acc[r] + bs;
        if (act == 1) v = gelu_f(v);
        if (res) v += res[(size_t)Mo * Nc + No];
        out[(size_t)Mo * Nc + No] = v;
    }
}

// ---------------------------------------------------------------------------
// Slice-weight softmax: sw[n,h,g] = softmax_g((xm[n,h,:4] @ slw + slb)/temp[h])
// ---------------------------------------------------------------------------
__global__ void k_slice_softmax(const float* __restrict__ xm, const float* __restrict__ slw,
                                const float* __restrict__ slb, const float* __restrict__ temp,
                                float* __restrict__ sw, int N) {
    int t = blockIdx.x * blockDim.x + threadIdx.x;
    if (t >= N * 8) return;
    int n = t >> 3, hh = t & 7;
    float x0 = xm[n * 32 + hh * 4 + 0], x1 = xm[n * 32 + hh * 4 + 1];
    float x2 = xm[n * 32 + hh * 4 + 2], x3 = xm[n * 32 + hh * 4 + 3];
    float tinv = 1.0f / temp[hh];
    float l[32]; float mx = -3.0e38f;
    #pragma unroll
    for (int g = 0; g < 32; ++g) {
        float v = slb[g];
        v = fmaf(x0, slw[0 * 32 + g], v);
        v = fmaf(x1, slw[1 * 32 + g], v);
        v = fmaf(x2, slw[2 * 32 + g], v);
        v = fmaf(x3, slw[3 * 32 + g], v);
        v *= tinv;
        l[g] = v;
        mx = fmaxf(mx, v);
    }
    float s = 0.0f;
    #pragma unroll
    for (int g = 0; g < 32; ++g) { float e = expf(l[g] - mx); l[g] = e; s += e; }
    float r = 1.0f / s;
    #pragma unroll
    for (int g = 0; g < 32; ++g) sw[(size_t)n * 256 + hh * 32 + g] = l[g] * r;
}

// ---------------------------------------------------------------------------
// Slice tokens with fused norm: stok[h,g,c] = (sum_n fx[n,h,c]*sw[n,h,g]) / (sum_n sw + 1e-5)
// One wave per (h,g).
// ---------------------------------------------------------------------------
__global__ void k_stok(const float* __restrict__ fx, const float* __restrict__ sw,
                       float* __restrict__ stok, int N) {
    int w = (blockIdx.x * blockDim.x + threadIdx.x) >> 5;  // 0..255
    int lane = threadIdx.x & 31;
    int hh = w >> 5, g = w & 31;
    float s = 0.0f, c0 = 0.0f, c1 = 0.0f, c2 = 0.0f, c3 = 0.0f;
    for (int n = lane; n < N; n += WAVE) {
        float sv = sw[(size_t)n * 256 + hh * 32 + g];
        const float* f = fx + (size_t)n * 32 + hh * 4;
        s += sv;
        c0 = fmaf(sv, f[0], c0);
        c1 = fmaf(sv, f[1], c1);
        c2 = fmaf(sv, f[2], c2);
        c3 = fmaf(sv, f[3], c3);
    }
    s  = wred_sum(s);
    c0 = wred_sum(c0); c1 = wred_sum(c1); c2 = wred_sum(c2); c3 = wred_sum(c3);
    if (lane == 0) {
        float r = 1.0f / (s + 1e-5f);
        stok[hh * 128 + g * 4 + 0] = c0 * r;
        stok[hh * 128 + g * 4 + 1] = c1 * r;
        stok[hh * 128 + g * 4 + 2] = c2 * r;
        stok[hh * 128 + g * 4 + 3] = c3 * r;
    }
}

// ---------------------------------------------------------------------------
// q,k,v = stok @ wq/wk/wv   (per head, 4x4 weights, no bias)
// ---------------------------------------------------------------------------
__global__ void k_qkv(const float* __restrict__ stok,
                      const float* __restrict__ wq, const float* __restrict__ wk,
                      const float* __restrict__ wv,
                      float* __restrict__ qb, float* __restrict__ kb, float* __restrict__ vb) {
    int t = blockIdx.x * blockDim.x + threadIdx.x;
    if (t >= 1024) return;
    int hh = t >> 7, g = (t >> 2) & 31, c = t & 3;
    const float* sr = stok + hh * 128 + g * 4;
    float aq = 0.0f, ak = 0.0f, av = 0.0f;
    #pragma unroll
    for (int cp = 0; cp < 4; ++cp) {
        float sv = sr[cp];
        aq = fmaf(sv, wq[cp * 4 + c], aq);
        ak = fmaf(sv, wk[cp * 4 + c], ak);
        av = fmaf(sv, wv[cp * 4 + c], av);
    }
    qb[t] = aq; kb[t] = ak; vb[t] = av;
}

// ---------------------------------------------------------------------------
// Per-head token attention: one wave per (h,g); lane = f.
// o[h,g,:] = softmax_f(q[g]·k[f]*0.5) @ v
// ---------------------------------------------------------------------------
__global__ void k_attn(const float* __restrict__ qb, const float* __restrict__ kb,
                       const float* __restrict__ vb, float* __restrict__ ob) {
    int w = (blockIdx.x * blockDim.x + threadIdx.x) >> 5;  // 0..255
    int lane = threadIdx.x & 31;
    int hh = w >> 5, g = w & 31;
    float q0 = qb[hh * 128 + g * 4 + 0], q1 = qb[hh * 128 + g * 4 + 1];
    float q2 = qb[hh * 128 + g * 4 + 2], q3 = qb[hh * 128 + g * 4 + 3];
    const float* kr = kb + hh * 128 + lane * 4;
    float logit = 0.5f * (q0 * kr[0] + q1 * kr[1] + q2 * kr[2] + q3 * kr[3]);
    float m = wred_max(logit);
    float e = expf(logit - m);
    float s = wred_sum(e);
    float a = e / s;
    const float* vr = vb + hh * 128 + lane * 4;
    float o0 = wred_sum(a * vr[0]);
    float o1 = wred_sum(a * vr[1]);
    float o2 = wred_sum(a * vr[2]);
    float o3 = wred_sum(a * vr[3]);
    if (lane == 0) {
        ob[hh * 128 + g * 4 + 0] = o0;
        ob[hh * 128 + g * 4 + 1] = o1;
        ob[hh * 128 + g * 4 + 2] = o2;
        ob[hh * 128 + g * 4 + 3] = o3;
    }
}

// ---------------------------------------------------------------------------
// De-slice: mix[n, h*4+c] = sum_g o[h,g,c] * sw[n,h,g]
// ---------------------------------------------------------------------------
__global__ void k_mix(const float* __restrict__ ob, const float* __restrict__ sw,
                      float* __restrict__ mix, int N) {
    int t = blockIdx.x * blockDim.x + threadIdx.x;
    if (t >= N * 32) return;
    int n = t >> 5, hc = t & 31;
    int hh = hc >> 2, c = hc & 3;
    const float* swr = sw + (size_t)n * 256 + hh * 32;
    const float* obr = ob + hh * 128 + c;
    float v = 0.0f;
    #pragma unroll
    for (int g = 0; g < 32; ++g) v = fmaf(swr[g], obr[g * 4], v);
    mix[t] = v;
}

// ===========================================================================
// Host orchestration
// ===========================================================================
struct PAp {
    const float *fx_b, *fx_w, *out_b, *out_w, *sl_b, *sl_w, *temp,
                *wk, *wq, *wv, *x_b, *x_w;
};

static const int CI[12] = {3, 8, 16, 32, 32, 32, 32, 32, 32, 32, 16, 8};
static const int CO[12] = {8, 16, 32, 32, 32, 32, 32, 32, 32, 16, 8, 1};

extern "C" void kernel_launch(void* const* d_in, const int* in_sizes, int n_in,
                              void* d_out, int out_size, void* d_ws, size_t ws_size,
                              hipStream_t stream) {
    (void)n_in; (void)out_size; (void)ws_size;
    const float* x      = (const float*)d_in[0];
    const float* pos    = (const float*)d_in[1];
    const int*   ei     = (const int*)d_in[2];
    const float* pseudo = (const float*)d_in[3];
    const int N = in_sizes[0] / 3;
    const int E = in_sizes[2] / 2;
    const int* src = ei;
    const int* tgt = ei + E;

    // ---- params (top-level dict order; sorted-key tree flatten inside 'params')
    int id = 4;
    const float *bn_b[11], *bn_g[11], *bn_m[11], *bn_v[11];
    for (int i = 0; i < 11; ++i) {
        bn_b[i] = (const float*)d_in[id++]; bn_g[i] = (const float*)d_in[id++];
        bn_m[i] = (const float*)d_in[id++]; bn_v[i] = (const float*)d_in[id++];
    }
    const float *cv_bias[12], *cv_root[12], *cv_w[12];
    for (int i = 0; i < 12; ++i) {
        cv_bias[i] = (const float*)d_in[id++];
        cv_root[i] = (const float*)d_in[id++];
        cv_w[i]    = (const float*)d_in[id++];
    }
    const float* enc1_b = (const float*)d_in[id++]; const float* enc1_w = (const float*)d_in[id++];
    const float* enc2_b = (const float*)d_in[id++]; const float* enc2_w = (const float*)d_in[id++];
    const float* ep1_b  = (const float*)d_in[id++]; const float* ep1_w  = (const float*)d_in[id++];
    const float* ep2_b  = (const float*)d_in[id++]; const float* ep2_w  = (const float*)d_in[id++];
    const float* ln1_b  = (const float*)d_in[id++]; const float* ln1_g  = (const float*)d_in[id++];
    const float* ln2_b  = (const float*)d_in[id++]; const float* ln2_g  = (const float*)d_in[id++];
    const float* mlp1_b1 = (const float*)d_in[id++]; const float* mlp1_b2 = (const float*)d_in[id++];
    const float* mlp1_w1 = (const float*)d_in[id++]; const float* mlp1_w2 = (const float*)d_in[id++];
    const float* mlp2_b1 = (const float*)d_in[id++]; const float* mlp2_b2 = (const float*)d_in[id++];
    const float* mlp2_w1 = (const float*)d_in[id++]; const float* mlp2_w2 = (const float*)d_in[id++];
    PAp pa[2];
    for (int p = 0; p < 2; ++p) {
        pa[p].fx_b  = (const float*)d_in[id++]; pa[p].fx_w  = (const float*)d_in[id++];
        pa[p].out_b = (const float*)d_in[id++]; pa[p].out_w = (const float*)d_in[id++];
        pa[p].sl_b  = (const float*)d_in[id++]; pa[p].sl_w  = (const float*)d_in[id++];
        pa[p].temp  = (const float*)d_in[id++];
        pa[p].wk    = (const float*)d_in[id++]; pa[p].wq    = (const float*)d_in[id++];
        pa[p].wv    = (const float*)d_in[id++];
        pa[p].x_b   = (const float*)d_in[id++]; pa[p].x_w   = (const float*)d_in[id++];
    }

    // ---- workspace carve (floats, 64B-aligned chunks for b128 loads)
    float* W0 = (float*)d_ws;
    size_t off = 0;
    auto alloc = [&](size_t n) { float* p = W0 + off; off += (n + 15) & ~(size_t)15; return p; };
    float* agg   = alloc((size_t)N * 32);
    float* cnt   = alloc((size_t)N);
    float* degS  = alloc((size_t)N);
    float* ef    = alloc((size_t)N * 3);
    float* enc   = alloc((size_t)N * 4);
    float* hA    = alloc((size_t)N * 32);
    float* hB    = alloc((size_t)N * 32);
    float* hbv   = alloc((size_t)N * 32);
    float* lnb   = alloc((size_t)N * 32);
    float* fxb   = alloc((size_t)N * 32);
    float* xmb   = alloc((size_t)N * 32);
    float* mixb  = alloc((size_t)N * 32);
    float* swb   = alloc((size_t)N * 256);
    float* mlph  = alloc((size_t)N * 128);
    float* stokb = alloc(1024);
    float* qb    = alloc(1024);
    float* kbuf  = alloc(1024);
    float* vbuf  = alloc(1024);
    float* obuf  = alloc(1024);
    // packed f16 operands (sizes in halves / 2 -> floats)
    const int MtMax = (N + 15) / 16;
    _Float16* pA = (_Float16*)alloc((size_t)MtMax * 4 * 512 / 2 + 16);   // up to K=128
    _Float16* pB = (_Float16*)alloc(4096 / 2 + 16);                      // up to K*Nc = 4096

    // ---- preamble: degrees, edge features, encoder
    hipMemsetAsync(cnt,  0, (size_t)N * 4, stream);
    hipMemsetAsync(degS, 0, (size_t)N * 4, stream);
    k_count_deg<<<(E + 255) / 256, 256, 0, stream>>>(src, tgt, cnt, degS, E);
    k_edge_feats<<<(N + 7) / 8, 256, 0, stream>>>(pos, degS, ef, N);
    k_encoder<<<(N + 255) / 256, 256, 0, stream>>>(ef, enc1_w, enc1_b, enc2_w, enc2_b, enc, N);

    const float* hcur = x;
    float* pp[2] = {hA, hB};
    int tog = 0;

    auto conv = [&](int l, bool bn, float* outp) {
        int Ci = CI[l], Co = CO[l];
        hipMemsetAsync(agg, 0, (size_t)N * Co * 4, stream);
        k_spline_edge<<<(E + 7) / 8, 256, 0, stream>>>(hcur, pseudo, src, tgt, cv_w[l], agg, E, Ci, Co);
        k_node_update<<<((size_t)N * Co + 255) / 256, 256, 0, stream>>>(
            hcur, agg, cnt, cv_root[l], cv_bias[l],
            bn ? bn_g[l] : nullptr, bn ? bn_b[l] : nullptr,
            bn ? bn_m[l] : nullptr, bn ? bn_v[l] : nullptr,
            outp, N, Ci, Co);
        hcur = outp;
    };

    auto pack_a = [&](const float* A, int M, int K) {
        int total = ((M + 15) / 16) * (K / 32) * 512;
        k_pack_a<<<(total + 255) / 256, 256, 0, stream>>>(A, pA, M, K, total);
    };
    auto wmma_lin = [&](const float* Wt, const float* bias, const float* res,
                        float* out, int M, int K, int Nc, int act) {
        int totalB = (Nc / 16) * (K / 32) * 512;
        k_pack_b<<<(totalB + 255) / 256, 256, 0, stream>>>(Wt, pB, K, Nc, totalB);
        dim3 grid((unsigned)((M + 127) / 128), (unsigned)(Nc / 16));
        k_wmma_linear<<<grid, 256, 0, stream>>>(pA, pB, bias, res, out, M, K / 32, Nc, act);
    };

    auto attn_block = [&](const PAp& p, const float* epw, const float* epb,
                          const float* lng, const float* lnbb,
                          const float* w1, const float* b1,
                          const float* w2, const float* b2) {
        k_add_ep<<<((size_t)N * 32 + 255) / 256, 256, 0, stream>>>(hcur, enc, epw, epb, hbv, N);
        k_layernorm<<<(N + 255) / 256, 256, 0, stream>>>(hbv, lng, lnbb, lnb, N);
        pack_a(lnb, N, 32);                                        // shared by fx and xm
        wmma_lin(p.fx_w, p.fx_b, nullptr, fxb, N, 32, 32, 0);
        wmma_lin(p.x_w,  p.x_b,  nullptr, xmb, N, 32, 32, 0);
        k_slice_softmax<<<((size_t)N * 8 + 255) / 256, 256, 0, stream>>>(xmb, p.sl_w, p.sl_b, p.temp, swb, N);
        k_stok<<<32, 256, 0, stream>>>(fxb, swb, stokb, N);
        k_qkv<<<4, 256, 0, stream>>>(stokb, p.wq, p.wk, p.wv, qb, kbuf, vbuf);
        k_attn<<<32, 256, 0, stream>>>(qb, kbuf, vbuf, obuf);
        k_mix<<<((size_t)N * 32 + 255) / 256, 256, 0, stream>>>(obuf, swb, mixb, N);
        pack_a(mixb, N, 32);
        wmma_lin(p.out_w, p.out_b, hbv, hbv, N, 32, 32, 0);        // attn out-proj + residual
        pack_a(hbv, N, 32);
        wmma_lin(w1, b1, nullptr, mlph, N, 32, 128, 1);            // MLP up + GELU
        pack_a(mlph, N, 128);
        wmma_lin(w2, b2, hbv, hbv, N, 128, 32, 0);                 // MLP down + residual
        hcur = hbv;
    };

    for (int l = 0; l < 3; ++l) { conv(l, true, pp[tog]); tog ^= 1; }
    attn_block(pa[0], ep1_w, ep1_b, ln1_g, ln1_b, mlp1_w1, mlp1_b1, mlp1_w2, mlp1_b2);
    for (int l = 3; l < 6; ++l) { conv(l, true, pp[tog]); tog ^= 1; }
    attn_block(pa[1], ep2_w, ep2_b, ln2_g, ln2_b, mlp2_w1, mlp2_b1, mlp2_w2, mlp2_b2);
    for (int l = 6; l < 11; ++l) { conv(l, true, pp[tog]); tog ^= 1; }
    conv(11, false, (float*)d_out);   // final layer: Cout=1, ELU only, writes d_out[N]
}
